// SWINTransformerEncoder_31138512896312
// MI455X (gfx1250) — matrix operand order repaired and checked by reading, not compiled, not run on hardware
//
#include <hip/hip_runtime.h>

// ---------------- CDNA5 types ----------------
typedef __attribute__((ext_vector_type(16))) __bf16 v16bf;
typedef __attribute__((ext_vector_type(8)))  float  v8f;
typedef __attribute__((ext_vector_type(4)))  unsigned v4u;
typedef __attribute__((ext_vector_type(8)))  int      v8i;
typedef __attribute__((ext_vector_type(4)))  int      v4i;

union Frag16 { v16bf v; unsigned u[8]; };

__device__ __forceinline__ unsigned f32_to_bf16_bits(float f) {
  unsigned u = __float_as_uint(f);
  unsigned r = u + 0x7FFFu + ((u >> 16) & 1u);  // round-to-nearest-even
  return r >> 16;
}
__device__ __forceinline__ unsigned pack_bf16x2(float lo, float hi) {
  return f32_to_bf16_bits(lo) | (f32_to_bf16_bits(hi) << 16);
}
__device__ __forceinline__ float bf16_lo(unsigned u) { return __uint_as_float(u << 16); }
__device__ __forceinline__ float bf16_hi(unsigned u) { return __uint_as_float(u & 0xFFFF0000u); }
__device__ __forceinline__ float gelu_tanh(float x) {
  float x3 = x * x * x;
  return 0.5f * x * (1.0f + tanhf(0.7978845608028654f * (x + 0.044715f * x3)));
}

// ---------------- Tensor Data Mover: load [rows]x32 bf16 tile into padded LDS ----------------
// D# per CDNA5 ISA: data_size=2B, 2D tile, row stride k_elems,
// LDS padding 1 DWORD per 16 DWORDs (pad_interval=3 -> 16 dw, pad_amount=0 -> 1 dw)
// which reproduces the [rows][17] dword LDS layout used by the WMMA fragment loads.
__device__ __forceinline__ void tdm_load_tile_bf16(unsigned lds_byte_addr,
                                                   const unsigned short* gsrc,
                                                   int rows, int k_elems)
{
  unsigned long long ga = (unsigned long long)(size_t)gsrc;
  v4u g0;
  g0[0] = 1u;                                                  // count=1 (valid user D#)
  g0[1] = lds_byte_addr;                                       // lds_addr [63:32]
  g0[2] = (unsigned)ga;                                        // global_addr [95:64]
  g0[3] = (unsigned)((ga >> 32) & 0x01FFFFFFull) | (2u << 30); // addr[56:32] + type=2
  unsigned td0 = (unsigned)k_elems;                            // tensor_dim0 (elements)
  unsigned td1 = 0x7FFFFFFFu;                                  // tensor_dim1 (no OOB)
  v8i g1;
  g1[0] = (int)((1u << 16) | (1u << 20) | (3u << 22));         // data_size=2B|pad_en|interval=16dw, amount=1dw
  g1[1] = (int)((td0 & 0xFFFFu) << 16);                        // atomic_addr=0, tensor_dim0 lo16
  g1[2] = (int)((td0 >> 16) | ((td1 & 0xFFFFu) << 16));        // tensor_dim0 hi16, tensor_dim1 lo16
  g1[3] = (int)((td1 >> 16) | (32u << 16));                    // tensor_dim1 hi16, tile_dim0=32
  g1[4] = rows;                                                // tile_dim1=rows, tile_dim2=0 (2D)
  g1[5] = k_elems;                                             // tensor_dim0_stride lo32
  g1[6] = 0;                                                   // stride hi, dim1_stride lo
  g1[7] = 0;
  v4i z4 = {0, 0, 0, 0};
  v8i z8 = {0, 0, 0, 0, 0, 0, 0, 0};
  // 6-arg form on this toolchain: (g0, g1, g2, g3, g4, cpol)
  __builtin_amdgcn_tensor_load_to_lds(g0, g1, z4, z4, z8, 0);
}

// ---------------- bf16-WMMA GEMM with TDM staging: C = act(A@Bt^T + bias) + Res ----------------
// A:  [M,K] bf16 row-major.  Bt: [N,K] bf16 row-major (pre-transposed weight).
// C: f32 (outfmt=0) or bf16 (outfmt=1).  M%128==0, N%64==0, K%32==0 at all call sites.
__global__ __launch_bounds__(256)
void gemm_bf16_wmma(const unsigned short* __restrict__ A,
                    const unsigned short* __restrict__ Bt,
                    const float* __restrict__ bias, const float* __restrict__ Res,
                    void* __restrict__ Cout, int Msz, int Nsz, int Ksz,
                    int act, int outfmt)
{
  __shared__ unsigned As[2][128][17];   // bf16 pairs [row][kpair] + 1 dw pad (TDM-generated)
  __shared__ unsigned Bs[2][64][17];

  const int tid   = threadIdx.x;
  const int lane  = tid & 31;
  const int wave  = tid >> 5;            // 0..7
  const int waveM = wave >> 1;           // 0..3
  const int waveN = wave & 1;            // 0..1
  const int g     = lane >> 4;
  const int lm    = lane & 15;

  const int bm = blockIdx.y * 128;
  const int bn = blockIdx.x * 64;

  v8f acc[2][2];
  #pragma unroll
  for (int mt = 0; mt < 2; ++mt)
    #pragma unroll
    for (int nt = 0; nt < 2; ++nt)
      #pragma unroll
      for (int r = 0; r < 8; ++r) acc[mt][nt][r] = 0.0f;

  const int nk = Ksz >> 5;
  if (tid < 32) {   // one wave drives the TDM (TENSORcnt is per-wave)
    tdm_load_tile_bf16((unsigned)(size_t)&As[0][0][0], A + (size_t)bm * Ksz, 128, Ksz);
    tdm_load_tile_bf16((unsigned)(size_t)&Bs[0][0][0], Bt + (size_t)bn * Ksz, 64, Ksz);
  }

  for (int kt = 0; kt < nk; ++kt) {
    const int cur = kt & 1;
    const int nxt = cur ^ 1;
    if (tid < 32) {
      if (kt + 1 < nk) {    // prefetch next tiles, overlap DMA with WMMA
        tdm_load_tile_bf16((unsigned)(size_t)&As[nxt][0][0],
                           A + (size_t)bm * Ksz + (kt + 1) * 32, 128, Ksz);
        tdm_load_tile_bf16((unsigned)(size_t)&Bs[nxt][0][0],
                           Bt + (size_t)bn * Ksz + (kt + 1) * 32, 64, Ksz);
        __builtin_amdgcn_s_wait_tensorcnt(2);   // current pair complete, next pair in flight
      } else {
        __builtin_amdgcn_s_wait_tensorcnt(0);
      }
    }
    __syncthreads();

    // fragment loads in contiguous k order (k = 16g + s), same order for A and B
    Frag16 af[2], bfr[2];
    #pragma unroll
    for (int mt = 0; mt < 2; ++mt) {
      int m = waveM * 32 + mt * 16 + lm;
      #pragma unroll
      for (int v = 0; v < 8; ++v) {
        int kp = g * 8 + v;  // K = 16g+2v
        af[mt].u[v] = As[cur][m][kp];
      }
    }
    #pragma unroll
    for (int nt = 0; nt < 2; ++nt) {
      int n = waveN * 32 + nt * 16 + lm;
      #pragma unroll
      for (int v = 0; v < 8; ++v) bfr[nt].u[v] = Bs[cur][n][g * 8 + v];  // K = 16g+2v
    }
    #pragma unroll
    for (int mt = 0; mt < 2; ++mt)
      #pragma unroll
      for (int nt = 0; nt < 2; ++nt)
        acc[mt][nt] = __builtin_amdgcn_wmma_f32_16x16x32_bf16(
            false, af[mt].v, false, bfr[nt].v, (short)0, acc[mt][nt], false, false);
    __syncthreads();  // all waves done reading LDS[cur] before it is reloaded
  }

  // epilogue: D layout -> row = r + 8*g, col = lane%16
  #pragma unroll
  for (int mt = 0; mt < 2; ++mt) {
    #pragma unroll
    for (int nt = 0; nt < 2; ++nt) {
      int col = bn + waveN * 32 + nt * 16 + lm;
      #pragma unroll
      for (int r = 0; r < 8; ++r) {
        int row = bm + waveM * 32 + mt * 16 + r + 8 * g;
        float vv = acc[mt][nt][r];
        if (bias) vv += bias[col];
        if (act == 1) vv = gelu_tanh(vv);
        if (Res) vv += Res[(size_t)row * Nsz + col];
        if (outfmt == 0) ((float*)Cout)[(size_t)row * Nsz + col] = vv;
        else ((unsigned short*)Cout)[(size_t)row * Nsz + col] = (unsigned short)f32_to_bf16_bits(vv);
      }
    }
  }
}

// ---------------- weight transpose + bf16 convert: W[K,N] f32 -> Wt[N,K] bf16 ----------------
__global__ __launch_bounds__(256)
void transpose_w_bf16(const float* __restrict__ W, unsigned short* __restrict__ Wt,
                      int Kd, int Nd)
{
  int gid = blockIdx.x * 256 + threadIdx.x;
  if (gid >= Kd * Nd) return;
  int n = gid / Kd, k = gid % Kd;  // consecutive gid -> consecutive k: coalesced writes
  Wt[(size_t)n * Kd + k] = (unsigned short)f32_to_bf16_bits(W[(size_t)k * Nd + n]);
}

// ---------------- LayerNorm D=256, wave/token; f32 in, packed bf16 out ----------------
__global__ __launch_bounds__(256)
void layernorm_bf16_kernel(const float* __restrict__ x, const float* __restrict__ gamma,
                           const float* __restrict__ beta, unsigned short* __restrict__ y,
                           int ntok)
{
  int token = blockIdx.x * 8 + (threadIdx.x >> 5);
  int lane  = threadIdx.x & 31;
  if (token >= ntok) return;
  const float4* r4 = (const float4*)(x + (size_t)token * 256);
  float4 a = r4[lane * 2], b = r4[lane * 2 + 1];
  float v[8] = {a.x, a.y, a.z, a.w, b.x, b.y, b.z, b.w};
  float s = 0.f;
  #pragma unroll
  for (int i = 0; i < 8; ++i) s += v[i];
  #pragma unroll
  for (int o = 16; o > 0; o >>= 1) s += __shfl_xor(s, o, 32);
  float mu = s * (1.0f / 256.0f);
  float q = 0.f;
  #pragma unroll
  for (int i = 0; i < 8; ++i) { float d = v[i] - mu; q += d * d; }
  #pragma unroll
  for (int o = 16; o > 0; o >>= 1) q += __shfl_xor(q, o, 32);
  float rstd = rsqrtf(q * (1.0f / 256.0f) + 1e-5f);
  int c0 = lane * 8;
  float nv[8];
  #pragma unroll
  for (int i = 0; i < 8; ++i) nv[i] = gamma[c0 + i] * (v[i] - mu) * rstd + beta[c0 + i];
  uint4 o4 = make_uint4(pack_bf16x2(nv[0], nv[1]), pack_bf16x2(nv[2], nv[3]),
                        pack_bf16x2(nv[4], nv[5]), pack_bf16x2(nv[6], nv[7]));
  ((uint4*)(y + (size_t)token * 256))[lane] = o4;
}

// ---------------- fused gather zc[idx] + LayerNorm -> packed bf16 KV rows ----------------
__global__ __launch_bounds__(256)
void gather_layernorm_bf16_kernel(const float* __restrict__ zc, const int* __restrict__ idx,
                                  const float* __restrict__ gamma, const float* __restrict__ beta,
                                  unsigned short* __restrict__ y, int nrows)
{
  int rowi = blockIdx.x * 8 + (threadIdx.x >> 5);
  int lane = threadIdx.x & 31;
  if (rowi >= nrows) return;
  int m   = rowi / (2048 * 9);
  int src = idx[rowi];
  const float4* r4 = (const float4*)(zc + ((size_t)m * 4096 + src) * 256);
  float4 a = r4[lane * 2], b = r4[lane * 2 + 1];
  float v[8] = {a.x, a.y, a.z, a.w, b.x, b.y, b.z, b.w};
  float s = 0.f;
  #pragma unroll
  for (int i = 0; i < 8; ++i) s += v[i];
  #pragma unroll
  for (int o = 16; o > 0; o >>= 1) s += __shfl_xor(s, o, 32);
  float mu = s * (1.0f / 256.0f);
  float q = 0.f;
  #pragma unroll
  for (int i = 0; i < 8; ++i) { float d = v[i] - mu; q += d * d; }
  #pragma unroll
  for (int o = 16; o > 0; o >>= 1) q += __shfl_xor(q, o, 32);
  float rstd = rsqrtf(q * (1.0f / 256.0f) + 1e-5f);
  int c0 = lane * 8;
  float nv[8];
  #pragma unroll
  for (int i = 0; i < 8; ++i) nv[i] = gamma[c0 + i] * (v[i] - mu) * rstd + beta[c0 + i];
  uint4 o4 = make_uint4(pack_bf16x2(nv[0], nv[1]), pack_bf16x2(nv[2], nv[3]),
                        pack_bf16x2(nv[4], nv[5]), pack_bf16x2(nv[6], nv[7]));
  ((uint4*)(y + (size_t)rowi * 256))[lane] = o4;
}

// ---------------- top-K(9) nearest grid points per target (stable ties) ----------------
__global__ __launch_bounds__(256)
void topk_kernel(const float* __restrict__ xt, const float* __restrict__ xc,
                 int* __restrict__ idx)
{
  int gid = blockIdx.x * 256 + threadIdx.x;   // m*2048+n
  if (gid >= 4 * 2048) return;
  int m = gid >> 11;
  float tx = xt[(size_t)gid * 2 + 0];
  float ty = xt[(size_t)gid * 2 + 1];
  const float* gp = xc + (size_t)m * 4096 * 2;
  float bd[9]; int bi[9];
  #pragma unroll
  for (int i = 0; i < 9; ++i) { bd[i] = 3.4e38f; bi[i] = 0; }
  for (int c = 0; c < 4096; ++c) {
    float dx = gp[c * 2 + 0] - tx;
    float dy = gp[c * 2 + 1] - ty;
    float d = dx * dx + dy * dy;
    if (d < bd[8]) {                           // strict < keeps lower index first on ties
      int pos = 8;
      while (pos > 0 && bd[pos - 1] > d) {
        bd[pos] = bd[pos - 1]; bi[pos] = bi[pos - 1]; --pos;
      }
      bd[pos] = d; bi[pos] = c;
    }
  }
  #pragma unroll
  for (int i = 0; i < 9; ++i) idx[(size_t)gid * 9 + i] = bi[i];
}

// ---------------- windowed MHSA; qkv bf16 [rows,768]; out bf16 [rows,256] ----------------
__global__ __launch_bounds__(64)
void swin_attn_kernel(const unsigned short* __restrict__ qkv, unsigned short* __restrict__ out,
                      int shift)
{
  __shared__ float kbuf[64][32];
  __shared__ float vbuf[64][32];
  __shared__ float sbuf[64][64];
  int b = blockIdx.x;                 // 256 windows * 8 heads
  int head = b & 7;
  int w = b >> 3;
  int m = w >> 6;
  int widx = w & 63;
  int wi = widx >> 3, wj = widx & 7;
  int t = threadIdx.x;
  int ti = t >> 3, tj = t & 7;
  int ii = (wi * 8 + ti + shift) & 63;
  int jj = (wj * 8 + tj + shift) & 63;
  size_t row = (size_t)m * 4096 + (size_t)ii * 64 + jj;
  const unsigned* p32 = (const unsigned*)qkv + row * 384 + head * 16;
  float q[32];
  #pragma unroll
  for (int c2 = 0; c2 < 16; ++c2) {
    unsigned uq = p32[c2];
    unsigned uk = p32[128 + c2];
    unsigned uv = p32[256 + c2];
    q[2 * c2] = bf16_lo(uq);       q[2 * c2 + 1] = bf16_hi(uq);
    kbuf[t][2 * c2] = bf16_lo(uk); kbuf[t][2 * c2 + 1] = bf16_hi(uk);
    vbuf[t][2 * c2] = bf16_lo(uv); vbuf[t][2 * c2 + 1] = bf16_hi(uv);
  }
  __syncthreads();
  const float scale = 0.17677669529663687f;  // 32^-0.5
  float mx = -3.4e38f;
  for (int j2 = 0; j2 < 64; ++j2) {
    float s = 0.f;
    #pragma unroll
    for (int c = 0; c < 32; ++c) s += q[c] * kbuf[j2][c];
    s *= scale;
    sbuf[t][j2] = s;
    mx = fmaxf(mx, s);
  }
  float sum = 0.f;
  for (int j2 = 0; j2 < 64; ++j2) {
    float p = expf(sbuf[t][j2] - mx);
    sbuf[t][j2] = p; sum += p;
  }
  float inv = 1.0f / sum;
  float o[32];
  #pragma unroll
  for (int c = 0; c < 32; ++c) o[c] = 0.f;
  for (int j2 = 0; j2 < 64; ++j2) {
    float p = sbuf[t][j2];
    #pragma unroll
    for (int c = 0; c < 32; ++c) o[c] += p * vbuf[j2][c];
  }
  unsigned* od = (unsigned*)out + row * 128 + head * 16;
  #pragma unroll
  for (int c2 = 0; c2 < 16; ++c2)
    od[c2] = pack_bf16x2(o[2 * c2] * inv, o[2 * c2 + 1] * inv);
}

// ---------------- MHCA over K=9 neighbors; all operands bf16 ----------------
__global__ __launch_bounds__(256)
void mhca_attn_kernel(const unsigned short* __restrict__ q, const unsigned short* __restrict__ k,
                      const unsigned short* __restrict__ v, unsigned short* __restrict__ out)
{
  int gid = blockIdx.x * 256 + threadIdx.x;
  if (gid >= 8192 * 8) return;
  int row = gid >> 3;
  int head = gid & 7;
  const unsigned* qp = (const unsigned*)q + (size_t)row * 128 + head * 16;
  float qq[32];
  #pragma unroll
  for (int c2 = 0; c2 < 16; ++c2) {
    unsigned u = qp[c2];
    qq[2 * c2] = bf16_lo(u); qq[2 * c2 + 1] = bf16_hi(u);
  }
  const float scale = 0.17677669529663687f;
  float s[9]; float mx = -3.4e38f;
  for (int kk = 0; kk < 9; ++kk) {
    const unsigned* kp = (const unsigned*)k + ((size_t)row * 9 + kk) * 128 + head * 16;
    float d = 0.f;
    #pragma unroll
    for (int c2 = 0; c2 < 16; ++c2) {
      unsigned u = kp[c2];
      d += qq[2 * c2] * bf16_lo(u) + qq[2 * c2 + 1] * bf16_hi(u);
    }
    s[kk] = d * scale;
    mx = fmaxf(mx, s[kk]);
  }
  float sum = 0.f;
  for (int kk = 0; kk < 9; ++kk) { s[kk] = expf(s[kk] - mx); sum += s[kk]; }
  float inv = 1.0f / sum;
  float o[32];
  #pragma unroll
  for (int c = 0; c < 32; ++c) o[c] = 0.f;
  for (int kk = 0; kk < 9; ++kk) {
    const unsigned* vp = (const unsigned*)v + ((size_t)row * 9 + kk) * 128 + head * 16;
    float p = s[kk];
    #pragma unroll
    for (int c2 = 0; c2 < 16; ++c2) {
      unsigned u = vp[c2];
      o[2 * c2] += p * bf16_lo(u); o[2 * c2 + 1] += p * bf16_hi(u);
    }
  }
  unsigned* od = (unsigned*)out + (size_t)row * 128 + head * 16;
  #pragma unroll
  for (int c2 = 0; c2 < 16; ++c2)
    od[c2] = pack_bf16x2(o[2 * c2] * inv, o[2 * c2 + 1] * inv);
}

// ---------------- host orchestration ----------------
extern "C" void kernel_launch(void* const* d_in, const int* in_sizes, int n_in,
                              void* d_out, int out_size, void* d_ws, size_t ws_size,
                              hipStream_t stream)
{
  (void)in_sizes; (void)n_in; (void)out_size; (void)ws_size;
  const float* xc    = (const float*)d_in[0];
  const float* zc0   = (const float*)d_in[1];
  const float* xt    = (const float*)d_in[2];
  const float* zt0   = (const float*)d_in[3];
  const float* s_ln1s = (const float*)d_in[4];
  const float* s_ln1b = (const float*)d_in[5];
  const float* s_Wqkv = (const float*)d_in[6];
  const float* s_Wo   = (const float*)d_in[7];
  const float* s_ln2s = (const float*)d_in[8];
  const float* s_ln2b = (const float*)d_in[9];
  const float* s_W1   = (const float*)d_in[10];
  const float* s_b1   = (const float*)d_in[11];
  const float* s_W2   = (const float*)d_in[12];
  const float* s_b2   = (const float*)d_in[13];
  const float* c_lnqs = (const float*)d_in[14];
  const float* c_lnqb = (const float*)d_in[15];
  const float* c_lnks = (const float*)d_in[16];
  const float* c_lnkb = (const float*)d_in[17];
  const float* c_Wq   = (const float*)d_in[18];
  const float* c_Wk   = (const float*)d_in[19];
  const float* c_Wv   = (const float*)d_in[20];
  const float* c_Wo   = (const float*)d_in[21];
  const float* c_ln2s = (const float*)d_in[22];
  const float* c_ln2b = (const float*)d_in[23];
  const float* c_W1   = (const float*)d_in[24];
  const float* c_b1   = (const float*)d_in[25];
  const float* c_W2   = (const float*)d_in[26];
  const float* c_b2   = (const float*)d_in[27];

  char* ws = (char*)d_ws;
  size_t off = 0;
  auto allocb = [&](size_t nbytes) { void* p = (void*)(ws + off); off += (nbytes + 255) & ~size_t(255); return p; };

  float* zc_buf = (float*)allocb(4ull * 4096 * 256 * 4);   // mutable grid features (f32)
  float* zt_buf = (float*)allocb(4ull * 2048 * 256 * 4);   // mutable target features (f32)
  int*   idx    = (int*)allocb(4ull * 2048 * 9 * 4);
  unsigned short* warena = (unsigned short*)allocb(2ull * 1048576 * 2);  // bf16 [N,K] weights
  unsigned short* R1 = (unsigned short*)allocb(18874368ull * 2); // qkv / kvn / attq+ffq (bf16)
  unsigned short* R2 = (unsigned short*)allocb(18874368ull * 2); // ff   / K-proj       (bf16)
  unsigned short* R3 = (unsigned short*)allocb(18874368ull * 2); // attn / V-proj       (bf16)
  unsigned short* R4 = (unsigned short*)allocb(4194304ull * 2);  // LN scratch / hq+qb  (bf16)

  (void)hipMemcpyAsync(zc_buf, zc0, 4ull * 4096 * 256 * 4, hipMemcpyDeviceToDevice, stream);
  (void)hipMemcpyAsync(zt_buf, zt0, 4ull * 2048 * 256 * 4, hipMemcpyDeviceToDevice, stream);

  topk_kernel<<<32, 256, 0, stream>>>(xt, xc, idx);

  // per-layer bf16 [N,K] weight views inside warena
  auto wt = [&](int l, size_t o) { return warena + (size_t)l * 1048576 + o; };
  for (int l = 0; l < 2; ++l) {
    transpose_w_bf16<<<768, 256, 0, stream>>>(s_Wqkv + (size_t)l * 256 * 768, wt(l, 0),      256, 768);
    transpose_w_bf16<<<256, 256, 0, stream>>>(s_Wo   + (size_t)l * 256 * 256, wt(l, 196608), 256, 256);
    transpose_w_bf16<<<512, 256, 0, stream>>>(s_W1   + (size_t)l * 256 * 512, wt(l, 262144), 256, 512);
    transpose_w_bf16<<<512, 256, 0, stream>>>(s_W2   + (size_t)l * 512 * 256, wt(l, 393216), 512, 256);
    transpose_w_bf16<<<256, 256, 0, stream>>>(c_Wq   + (size_t)l * 256 * 256, wt(l, 524288), 256, 256);
    transpose_w_bf16<<<256, 256, 0, stream>>>(c_Wk   + (size_t)l * 256 * 256, wt(l, 589824), 256, 256);
    transpose_w_bf16<<<256, 256, 0, stream>>>(c_Wv   + (size_t)l * 256 * 256, wt(l, 655360), 256, 256);
    transpose_w_bf16<<<256, 256, 0, stream>>>(c_Wo   + (size_t)l * 256 * 256, wt(l, 720896), 256, 256);
    transpose_w_bf16<<<512, 256, 0, stream>>>(c_W1   + (size_t)l * 256 * 512, wt(l, 786432), 256, 512);
    transpose_w_bf16<<<512, 256, 0, stream>>>(c_W2   + (size_t)l * 512 * 256, wt(l, 917504), 512, 256);
  }

  for (int l = 0; l < 2; ++l) {
    // ---- Swin layer ----
    layernorm_bf16_kernel<<<2048, 256, 0, stream>>>(zc_buf, s_ln1s + l * 256, s_ln1b + l * 256, R4, 16384);
    gemm_bf16_wmma<<<dim3(12, 128), 256, 0, stream>>>(R4, wt(l, 0), nullptr, nullptr,
                                                      R1, 16384, 768, 256, 0, 1);
    int shift = (l & 1) ? 4 : 0;
    swin_attn_kernel<<<2048, 64, 0, stream>>>(R1, R3, shift);
    gemm_bf16_wmma<<<dim3(4, 128), 256, 0, stream>>>(R3, wt(l, 196608), nullptr, zc_buf,
                                                     zc_buf, 16384, 256, 256, 0, 0);
    layernorm_bf16_kernel<<<2048, 256, 0, stream>>>(zc_buf, s_ln2s + l * 256, s_ln2b + l * 256, R4, 16384);
    gemm_bf16_wmma<<<dim3(8, 128), 256, 0, stream>>>(R4, wt(l, 262144), s_b1 + l * 512, nullptr,
                                                     R2, 16384, 512, 256, 1, 1);
    gemm_bf16_wmma<<<dim3(4, 128), 256, 0, stream>>>(R2, wt(l, 393216), s_b2 + l * 256, zc_buf,
                                                     zc_buf, 16384, 256, 512, 0, 0);

    // ---- MHCA layer ----
    unsigned short* hq = R4;
    unsigned short* qb = R4 + 2097152;
    layernorm_bf16_kernel<<<1024, 256, 0, stream>>>(zt_buf, c_lnqs + l * 256, c_lnqb + l * 256, hq, 8192);
    gemm_bf16_wmma<<<dim3(4, 64), 256, 0, stream>>>(hq, wt(l, 524288), nullptr, nullptr,
                                                    qb, 8192, 256, 256, 0, 1);
    gather_layernorm_bf16_kernel<<<9216, 256, 0, stream>>>(zc_buf, idx, c_lnks + l * 256, c_lnkb + l * 256,
                                                           R1, 73728);
    gemm_bf16_wmma<<<dim3(4, 576), 256, 0, stream>>>(R1, wt(l, 589824), nullptr, nullptr,
                                                     R2, 73728, 256, 256, 0, 1);
    gemm_bf16_wmma<<<dim3(4, 576), 256, 0, stream>>>(R1, wt(l, 655360), nullptr, nullptr,
                                                     R3, 73728, 256, 256, 0, 1);
    unsigned short* attq = R1;                 // kvn dead after K/V projections
    unsigned short* ffq  = R1 + 2097152;
    mhca_attn_kernel<<<256, 256, 0, stream>>>(qb, R2, R3, attq);
    gemm_bf16_wmma<<<dim3(4, 64), 256, 0, stream>>>(attq, wt(l, 720896), nullptr, zt_buf,
                                                    zt_buf, 8192, 256, 256, 0, 0);
    layernorm_bf16_kernel<<<1024, 256, 0, stream>>>(zt_buf, c_ln2s + l * 256, c_ln2b + l * 256, hq, 8192);
    gemm_bf16_wmma<<<dim3(8, 64), 256, 0, stream>>>(hq, wt(l, 786432), c_b1 + l * 512, nullptr,
                                                    ffq, 8192, 512, 256, 1, 1);
    float* final_out = (l == 1) ? (float*)d_out : zt_buf;
    gemm_bf16_wmma<<<dim3(4, 64), 256, 0, stream>>>(ffq, wt(l, 917504), c_b2 + l * 256, zt_buf,
                                                    final_out, 8192, 256, 512, 0, 0);
  }
}
